// CrossAttention_34772055228573
// MI455X (gfx1250) — compile-verified, matrix operand order checked by reading, not run on hardware
//
#include <hip/hip_runtime.h>

// ---------------------------------------------------------------------------
// Types for CDNA5 WMMA (wave32): v_wmma_f32_16x16x32_bf16
// ---------------------------------------------------------------------------
typedef __bf16 bf16_t;
typedef __attribute__((ext_vector_type(16))) __bf16 v16bf;
typedef __attribute__((ext_vector_type(8)))  __bf16 v8bf;
typedef __attribute__((ext_vector_type(8)))  float  v8f;
typedef int v4i __attribute__((vector_size(16)));   // matches builtin param type

#define AS1 __attribute__((address_space(1)))
#define AS3 __attribute__((address_space(3)))

#if __has_builtin(__builtin_amdgcn_global_load_async_to_lds_b128)
#define HAS_ASYNC_LDS 1
#else
#define HAS_ASYNC_LDS 0
#endif

// 16-byte global -> LDS copy; async (ASYNCcnt-tracked) when available.
__device__ __forceinline__ void copy16_g2l(const void* g, void* l) {
#if HAS_ASYNC_LDS
    __builtin_amdgcn_global_load_async_to_lds_b128((AS1 v4i*)g, (AS3 v4i*)l, 0, 0);
#else
    *(v8bf*)l = *(const v8bf*)g;
#endif
}

__device__ __forceinline__ void wait_async_lds() {
#if HAS_ASYNC_LDS
#if __has_builtin(__builtin_amdgcn_s_wait_asynccnt)
    __builtin_amdgcn_s_wait_asynccnt(0);
#else
    asm volatile("s_wait_asynccnt 0x0" ::: "memory");
#endif
#endif
}

__device__ __forceinline__ v8f wmma_bf16(v16bf a, v16bf b, v8f c) {
    // (neg_a, A, neg_b, B, c_mod, C, reuse_a, reuse_b)
    return __builtin_amdgcn_wmma_f32_16x16x32_bf16(false, a, false, b,
                                                   (short)0, c, false, false);
}

// Load a 16x32 bf16 fragment row (A-matrix row m, or transposed-B row n).
// CDNA5 16-bit A 16x32 layout: lane l holds row (l&15);
//   elements 0..7  -> K = (l>>4)*8 + e        (one 16B ds_load_b128)
//   elements 8..15 -> K = 16 + (l>>4)*8 + e-8 (one 16B ds_load_b128)
__device__ __forceinline__ v16bf load_frag(const bf16_t* __restrict__ row, int hi) {
    v8bf lo = *(const v8bf*)(row + hi * 8);
    v8bf hv = *(const v8bf*)(row + 16 + hi * 8);
    v16bf r;
#pragma unroll
    for (int i = 0; i < 8; ++i) { r[i] = lo[i]; r[8 + i] = hv[i]; }
    return r;
}

// ---------------------------------------------------------------------------
// Generic GEMM:  Y[M,N] = ( X[M,K] * W[K,N] + bias[N] ) * out_scale
// X either f32 (converted to bf16 on load) or bf16; Y either bf16 or f32.
// Block tile 128x128, BK=32, 256 threads = 8 waves, each wave 2x4 WMMA tiles.
// ---------------------------------------------------------------------------
template <bool IN_F32, bool OUT_BF16>
__global__ __launch_bounds__(256) void gemm_kernel(
    const void* __restrict__ Xv, const float* __restrict__ W,
    const float* __restrict__ bias, void* __restrict__ Yv,
    int M, int N, int K, float out_scale)
{
    __shared__ __align__(16) bf16_t As[128][40];  // 128 rows x 32 K (+8 pad)
    __shared__ __align__(16) bf16_t Bt[128][40];  // 128 cols(N) x 32 K (+8 pad)

    const int tid  = threadIdx.x;
    const int lane = tid & 31;
    const int wid  = tid >> 5;
    const int hi   = lane >> 4;
    const int col  = lane & 15;
    const int m0   = blockIdx.y * 128;
    const int n0   = blockIdx.x * 128;
    const int wr   = wid >> 1;   // 0..3 : 32-row slice
    const int wc   = wid & 1;    // 0..1 : 64-col slice

    const float*  Xf = (const float*)Xv;
    const bf16_t* Xb = (const bf16_t*)Xv;

    v8f acc[2][4] = {};

    const int arow = tid >> 1;          // 0..127
    const int acol = (tid & 1) * 16;    // 0 / 16
    const int bkk  = tid >> 3;          // 0..31  (K index)
    const int bn0  = (tid & 7) * 16;    // 0..112 (N index)

    for (int k0 = 0; k0 < K; k0 += 32) {
        // --- A tile: 128x32, 16 contiguous elements per thread ---
        if (IN_F32) {
            const float* src = Xf + (size_t)(m0 + arow) * K + k0 + acol;
            if (k0 + 32 < K) __builtin_prefetch(src + 32, 0, 0);
#pragma unroll
            for (int i = 0; i < 16; i += 4) {
                float4 f = *(const float4*)(src + i);
                As[arow][acol + i + 0] = (bf16_t)f.x;
                As[arow][acol + i + 1] = (bf16_t)f.y;
                As[arow][acol + i + 2] = (bf16_t)f.z;
                As[arow][acol + i + 3] = (bf16_t)f.w;
            }
        } else {
            const bf16_t* src = Xb + (size_t)(m0 + arow) * K + k0 + acol;
            if (k0 + 32 < K) __builtin_prefetch(src + 32, 0, 0);
            copy16_g2l(src,     &As[arow][acol]);
            copy16_g2l(src + 8, &As[arow][acol + 8]);
        }
        // --- B tile: W[k0..k0+32][n0..n0+128] stored K-transposed ---
        {
            const float* wsrc = W + (size_t)(k0 + bkk) * N + n0 + bn0;
            if (k0 + 32 < K) __builtin_prefetch(wsrc + (size_t)32 * N, 0, 0);
#pragma unroll
            for (int i = 0; i < 16; i += 4) {
                float4 f = *(const float4*)(wsrc + i);
                Bt[bn0 + i + 0][bkk] = (bf16_t)f.x;
                Bt[bn0 + i + 1][bkk] = (bf16_t)f.y;
                Bt[bn0 + i + 2][bkk] = (bf16_t)f.z;
                Bt[bn0 + i + 3][bkk] = (bf16_t)f.w;
            }
        }
        if (!IN_F32) wait_async_lds();
        __syncthreads();

        v16bf a0 = load_frag(&As[wr * 32 + col][0], hi);
        v16bf a1 = load_frag(&As[wr * 32 + 16 + col][0], hi);
        v16bf bfr[4];
#pragma unroll
        for (int ni = 0; ni < 4; ++ni)
            bfr[ni] = load_frag(&Bt[wc * 64 + ni * 16 + col][0], hi);
#pragma unroll
        for (int ni = 0; ni < 4; ++ni) {
            acc[0][ni] = wmma_bf16(a0, bfr[ni], acc[0][ni]);
            acc[1][ni] = wmma_bf16(a1, bfr[ni], acc[1][ni]);
        }
        __syncthreads();
    }

    // epilogue: C/D layout m = r + 8*hi, n = lane&15 within each 16x16 tile
#pragma unroll
    for (int mi = 0; mi < 2; ++mi)
#pragma unroll
    for (int ni = 0; ni < 4; ++ni)
#pragma unroll
    for (int r = 0; r < 8; ++r) {
        int m = m0 + wr * 32 + mi * 16 + r + 8 * hi;
        int n = n0 + wc * 64 + ni * 16 + col;
        float v = (acc[mi][ni][r] + bias[n]) * out_scale;
        if (OUT_BF16) ((bf16_t*)Yv)[(size_t)m * N + n] = (bf16_t)v;
        else          ((float*)Yv)[(size_t)m * N + n]  = v;
    }
}

// ---------------------------------------------------------------------------
// Flash attention: one block per (b, h, 64-row q tile). 256 threads / 8 waves.
// Q (pre-scaled bf16), K, V bf16 laid out [B][S][H*64]. bias f32 [H][Sq][Skv].
// ---------------------------------------------------------------------------
__global__ __launch_bounds__(256) void attn_kernel(
    const bf16_t* __restrict__ Qb, const bf16_t* __restrict__ Kb,
    const bf16_t* __restrict__ Vb, const float* __restrict__ bias,
    bf16_t* __restrict__ AOb)
{
    constexpr int S = 1024, DM = 1024;
    __shared__ __align__(16) bf16_t Qs[64][72];   // q rows x dk
    __shared__ __align__(16) bf16_t Ks[64][72];   // kv rows x dk (Bt for QK^T)
    __shared__ __align__(16) bf16_t Vt[64][72];   // dk rows x kv (Bt for PV)
    __shared__ __align__(16) bf16_t Ps[64][72];   // q rows x kv (bf16 probs)
    __shared__ float Sf[64][65];                  // f32 logits (padded)
    __shared__ float mrow[64], lrow[64], frow[64];

    const int tid  = threadIdx.x;
    const int lane = tid & 31;
    const int wid  = tid >> 5;
    const int hi   = lane >> 4;
    const int col  = lane & 15;

    const int bid = blockIdx.x;
    const int qt  = bid & 15;          // Sq/64 = 16
    const int h   = (bid >> 4) & 15;   // H = 16
    const int b   = bid >> 8;          // B = 4
    const int q0  = qt * 64;

    const size_t baseQ  = ((size_t)b * S + q0) * DM + h * 64;
    const size_t baseKV = (size_t)b * S * DM + h * 64;
    const float* biasBase = bias + ((size_t)h * S + q0) * S;

    const int ldr  = tid >> 2;         // 0..63
    const int ldc0 = (tid & 3) * 16;   // 0,16,32,48

    // load Q tile (async when available)
    {
        const bf16_t* src = Qb + baseQ + (size_t)ldr * DM + ldc0;
        copy16_g2l(src,     &Qs[ldr][ldc0]);
        copy16_g2l(src + 8, &Qs[ldr][ldc0 + 8]);
    }
    if (tid < 64) { mrow[tid] = -__builtin_inff(); lrow[tid] = 0.0f; }
    wait_async_lds();

    v8f oacc[2] = {};
    const int t0 = wid * 2;

    for (int kt = 0; kt < 16; ++kt) {
        const int kv0 = kt * 64;
        __syncthreads();  // previous iteration done with Ks/Vt/Ps/Sf
        // K tile (row-major, async path) and V tile (transposed to [dk][kv])
        {
            const bf16_t* ksrc = Kb + baseKV + (size_t)(kv0 + ldr) * DM + ldc0;
            copy16_g2l(ksrc,     &Ks[ldr][ldc0]);
            copy16_g2l(ksrc + 8, &Ks[ldr][ldc0 + 8]);
            const bf16_t* vsrc = Vb + baseKV + (size_t)(kv0 + ldr) * DM + ldc0;
            v8bf v0 = *(const v8bf*)(vsrc);
            v8bf v1 = *(const v8bf*)(vsrc + 8);
#pragma unroll
            for (int i = 0; i < 8; ++i) {
                Vt[ldc0 + i][ldr]     = v0[i];
                Vt[ldc0 + 8 + i][ldr] = v1[i];
            }
            // prefetch next kv tile's bias rows (streams 64MB from HBM)
            if (kt + 1 < 16)
                __builtin_prefetch(biasBase + (size_t)ldr * S + kv0 + 64, 0, 0);
        }
        wait_async_lds();
        __syncthreads();

        // S = Q * K^T (+ bias); each wave owns 2 of the 16 16x16 tiles
#pragma unroll
        for (int ti = 0; ti < 2; ++ti) {
            const int t  = t0 + ti;
            const int tm = t >> 2, tn = t & 3;
            v8f s = {};
#pragma unroll
            for (int ks = 0; ks < 2; ++ks) {
                v16bf a  = load_frag(&Qs[tm * 16 + col][ks * 32], hi);
                v16bf bb = load_frag(&Ks[tn * 16 + col][ks * 32], hi);
                s = wmma_bf16(a, bb, s);
            }
            const int n = tn * 16 + col;
#pragma unroll
            for (int r = 0; r < 8; ++r) {
                int m = tm * 16 + r + 8 * hi;
                Sf[m][n] = s[r] + biasBase[(size_t)m * S + kv0 + n];
            }
        }
        __syncthreads();

        // online softmax: one thread per q row
        if (tid < 64) {
            float mo = mrow[tid], mx = mo;
#pragma unroll 8
            for (int j = 0; j < 64; ++j) mx = fmaxf(mx, Sf[tid][j]);
            float fac = __expf(mo - mx);   // 0 on first tile (mo = -inf)
            float sum = 0.0f;
#pragma unroll 8
            for (int j = 0; j < 64; ++j) {
                float p = __expf(Sf[tid][j] - mx);
                sum += p;
                Ps[tid][j] = (bf16_t)p;
            }
            mrow[tid] = mx;
            lrow[tid] = lrow[tid] * fac + sum;
            frow[tid] = fac;
        }
        __syncthreads();

        // rescale O then O += P * V
#pragma unroll
        for (int ti = 0; ti < 2; ++ti) {
            const int t  = t0 + ti;
            const int tm = t >> 2, tn = t & 3;
#pragma unroll
            for (int r = 0; r < 8; ++r)
                oacc[ti][r] *= frow[tm * 16 + r + 8 * hi];
#pragma unroll
            for (int ks = 0; ks < 2; ++ks) {
                v16bf a  = load_frag(&Ps[tm * 16 + col][ks * 32], hi);
                v16bf bb = load_frag(&Vt[tn * 16 + col][ks * 32], hi);
                oacc[ti] = wmma_bf16(a, bb, oacc[ti]);
            }
        }
    }

    __syncthreads();
    if (tid < 64) {
        float l = lrow[tid];
        frow[tid] = (l > 0.0f) ? 1.0f / l : 0.0f;  // nan-zeroing analogue
    }
    __syncthreads();

#pragma unroll
    for (int ti = 0; ti < 2; ++ti) {
        const int t  = t0 + ti;
        const int tm = t >> 2, tn = t & 3;
#pragma unroll
        for (int r = 0; r < 8; ++r) {
            int m = tm * 16 + r + 8 * hi;
            int n = tn * 16 + col;
            float v = oacc[ti][r] * frow[m];
            AOb[baseQ + (size_t)m * DM + n] = (bf16_t)v;
        }
    }
}

// ---------------------------------------------------------------------------
// Launch
// ---------------------------------------------------------------------------
extern "C" void kernel_launch(void* const* d_in, const int* in_sizes, int n_in,
                              void* d_out, int out_size, void* d_ws, size_t ws_size,
                              hipStream_t stream)
{
    (void)in_sizes; (void)n_in; (void)out_size; (void)ws_size;
    const float* y1        = (const float*)d_in[0];
    const float* y2        = (const float*)d_in[1];
    const float* attn_bias = (const float*)d_in[2];
    const float* Wq = (const float*)d_in[3];
    const float* bq = (const float*)d_in[4];
    const float* Wk = (const float*)d_in[5];
    const float* bk = (const float*)d_in[6];
    const float* Wv = (const float*)d_in[7];
    const float* bv = (const float*)d_in[8];
    const float* Wo = (const float*)d_in[9];
    const float* bo = (const float*)d_in[10];
    float* out = (float*)d_out;

    const int B = 4, S = 1024, DM = 1024;
    const size_t elems = (size_t)B * S * DM;
    bf16_t* Qb = (bf16_t*)d_ws;
    bf16_t* Kb = Qb + elems;
    bf16_t* Vb = Kb + elems;
    bf16_t* AO = Vb + elems;

    const float SCALE = 0.125f;  // (DM/H)^-0.5 = 64^-0.5
    dim3 blk(256);
    dim3 grd(DM / 128, (B * S) / 128);

    // projections (f32 in -> bf16 out); Q pre-scaled by 1/sqrt(dk)
    gemm_kernel<true, true><<<grd, blk, 0, stream>>>(y2, Wq, bq, Qb, B * S, DM, DM, SCALE);
    gemm_kernel<true, true><<<grd, blk, 0, stream>>>(y1, Wk, bk, Kb, B * S, DM, DM, 1.0f);
    gemm_kernel<true, true><<<grd, blk, 0, stream>>>(y1, Wv, bv, Vb, B * S, DM, DM, 1.0f);

    // attention: B * H * (Sq/64) = 4 * 16 * 16 = 1024 blocks
    attn_kernel<<<dim3(1024), blk, 0, stream>>>(Qb, Kb, Vb, attn_bias, AO);

    // output projection (bf16 in -> f32 out)
    gemm_kernel<false, false><<<grd, blk, 0, stream>>>(AO, Wo, bo, out, B * S, DM, DM, 1.0f);
}
